// YOLOELoss_6811818131490
// MI455X (gfx1250) — compile-verified, matrix operand order checked by reading, not run on hardware
//
#include <hip/hip_runtime.h>
#include <hip/hip_bf16.h>
#include <math.h>

#define B_   32
#define A_   8400
#define G_   32
#define C_   80
#define RMAX 16
#define RB   17
#define TOPK 13

typedef __attribute__((ext_vector_type(16))) _Float16 v16h;
typedef __attribute__((ext_vector_type(8)))  float    v8f;

// ---- workspace layout (bytes) ----
#define OFF_BOXES 0ULL
#define OFF_S     (OFF_BOXES + 16ULL*B_*A_)   // float4 boxes
#define OFF_KEY   (OFF_S   + 4ULL*B_*A_)      // per-anchor BCE sums (reused as OHEM array)
#define OFF_FG    (OFF_KEY + 8ULL*B_*A_)      // packed (iou,31-g) keys
#define OFF_ACC   (OFF_FG  + 4ULL*B_*A_)      // fg flags
#define WS_NEEDED (OFF_ACC + 5ULL*B_*4ULL)    // npos,pos,box,dfl,neg (32 each)

__global__ void k_zero(unsigned* __restrict__ p, size_t n) {
  size_t i = (size_t)blockIdx.x * blockDim.x + threadIdx.x;
  if (i < n) p[i] = 0u;
}

// ---------- K1: DFL decode ----------
__global__ void k_decode(const float* __restrict__ reg,
                         const float* __restrict__ ap,
                         const float* __restrict__ st,
                         float4* __restrict__ boxes) {
  int idx = blockIdx.x * blockDim.x + threadIdx.x;
  if (idx >= B_ * A_) return;
  int a = idx % A_;
  const float* r = reg + (size_t)idx * (4 * RB);
  float d[4];
#pragma unroll
  for (int i = 0; i < 4; i++) {
    float m = -3.4e38f;
#pragma unroll
    for (int j = 0; j < RB; j++) m = fmaxf(m, r[i*RB + j]);
    float se = 0.f, dot = 0.f;
#pragma unroll
    for (int j = 0; j < RB; j++) { float e = expf(r[i*RB + j] - m); se += e; dot += e * (float)j; }
    d[i] = dot / se;
  }
  float ax = ap[2*a], ay = ap[2*a + 1], s = st[a];
  float4 b4;
  b4.x = ax - d[0]*s; b4.y = ay - d[1]*s;
  b4.z = ax + d[2]*s; b4.w = ay + d[3]*s;
  boxes[idx] = b4;
}

// ---------- K2: TAL assignment, one block per (b,g) ----------
__global__ void k_assign(const float* __restrict__ cls,
                         const float4* __restrict__ boxes,
                         const float* __restrict__ gtb,
                         const int* __restrict__ gtl,
                         const float* __restrict__ ap,
                         unsigned long long* __restrict__ keys,
                         unsigned* __restrict__ fg) {
  __shared__ float s_align[A_];
  __shared__ float s_val[256];
  __shared__ int   s_idx[256];
  __shared__ int   s_done;
  int b = blockIdx.x / G_, g = blockIdx.x % G_;
  int tid = threadIdx.x;
  const float* gb = gtb + ((size_t)b*G_ + g)*4;
  float gx1 = gb[0], gy1 = gb[1], gx2 = gb[2], gy2 = gb[3];
  float ga = fmaxf(gx2-gx1, 0.f) * fmaxf(gy2-gy1, 0.f);
  int lab = gtl[b*G_ + g];
  if (tid == 0) s_done = 0;
  for (int a = tid; a < A_; a += blockDim.x) {
    float4 p = boxes[(size_t)b*A_ + a];
    float iw = fmaxf(fminf(p.z, gx2) - fmaxf(p.x, gx1), 0.f);
    float ih = fmaxf(fminf(p.w, gy2) - fmaxf(p.y, gy1), 0.f);
    float inter = iw * ih;
    float pa = fmaxf(p.z-p.x, 0.f) * fmaxf(p.w-p.y, 0.f);
    float iou = inter / (pa + ga - inter + 1e-9f);
    float ax = ap[2*a], ay = ap[2*a + 1];
    bool ing = (ax > gx1) && (ax < gx2) && (ay > gy1) && (ay < gy2);
    float al = 0.f;
    if (ing) {
      float cs = fminf(fmaxf(cls[((size_t)b*A_ + a)*C_ + lab], 1e-7f), 1.f - 1e-7f);
      float i2 = iou*iou; float i6 = i2*i2*i2;
      al = sqrtf(cs) * i6 + 1e-11f;
    }
    s_align[a] = al;
  }
  __syncthreads();
  for (int t = 0; t < TOPK; t++) {
    float bv = -1e30f; int bi = 0x7fffffff;
    for (int a = tid; a < A_; a += blockDim.x) {
      float v = s_align[a];
      if (v > bv || (v == bv && a < bi)) { bv = v; bi = a; }
    }
    s_val[tid] = bv; s_idx[tid] = bi;
    __syncthreads();
    for (int off = 128; off > 0; off >>= 1) {
      if (tid < off) {
        float v2 = s_val[tid+off]; int i2 = s_idx[tid+off];
        if (v2 > s_val[tid] || (v2 == s_val[tid] && i2 < s_idx[tid])) { s_val[tid] = v2; s_idx[tid] = i2; }
      }
      __syncthreads();
    }
    if (tid == 0) {
      float v = s_val[0]; int ai = s_idx[0];
      if (v > 1e-11f) {
        float4 p = boxes[(size_t)b*A_ + ai];
        float iw = fmaxf(fminf(p.z, gx2) - fmaxf(p.x, gx1), 0.f);
        float ih = fmaxf(fminf(p.w, gy2) - fmaxf(p.y, gy1), 0.f);
        float inter = iw * ih;
        float pa = fmaxf(p.z-p.x, 0.f) * fmaxf(p.w-p.y, 0.f);
        float iou = inter / (pa + ga - inter + 1e-9f);
        unsigned long long key = ((unsigned long long)__float_as_uint(iou) << 32) | (unsigned)(31 - g);
        atomicMax(&keys[(size_t)b*A_ + ai], key);
        atomicOr(&fg[(size_t)b*A_ + ai], 1u);
        s_align[ai] = -1e30f;
      } else s_done = 1;
    }
    __syncthreads();
    if (s_done) break;
  }
}

// ---------- K3: per-anchor BCE class-sum via WMMA ----------
// Each wave processes 16-anchor x 80-class tiles. The 5120B tile is copied
// Global->LDS with 10 coalesced global_load_async_to_lds_b128 (ASYNCcnt),
// double-buffered so the next tile's DMA overlaps the current tile's
// f16 pack + v_wmma_f32_16x16x32_f16 reduction against an all-ones B.
#define TILE_F (16*C_)     // 1280 floats
__global__ void __launch_bounds__(256) k_bcesum(const float* __restrict__ cls,
                                                float* __restrict__ S) {
  __shared__ float s_tile[8][2][TILE_F];
  const int NT = (B_ * A_) / 16;
  int wid  = threadIdx.x >> 5;
  int lane = threadIdx.x & 31;
  int half = lane >> 4;
  int M    = lane & 15;
  int t0     = blockIdx.x * 8 + wid;
  int stride = gridDim.x * 8;

  // generic shared pointer -> LDS byte offset (low 32 bits of the address)
  unsigned lbase[2];
  lbase[0] = (unsigned)(size_t)(&s_tile[wid][0][0]);
  lbase[1] = (unsigned)(size_t)(&s_tile[wid][1][0]);

  v16h ones;
#pragma unroll
  for (int e = 0; e < 16; e++) ones[e] = (_Float16)1.0f;

  auto issue = [&](int tile, unsigned lb) {
    unsigned long long ga = (unsigned long long)(const void*)(cls + (size_t)tile * TILE_F);
#pragma unroll
    for (int k = 0; k < 10; k++) {
      unsigned loff = lb + (unsigned)(k * 512 + lane * 16);
      unsigned long long g = ga + (unsigned long long)(k * 512 + lane * 16);
      asm volatile("global_load_async_to_lds_b128 %0, %1, off"
                   :: "v"(loff), "v"(g) : "memory");
    }
  };

  if (t0 < NT) issue(t0, lbase[0]);
  int cur = 0;
  for (int t = t0; t < NT; t += stride) {
    int tn = t + stride;
    if (tn < NT) {
      issue(tn, lbase[cur ^ 1]);
      if (tn + stride < NT)
        __builtin_prefetch(cls + (size_t)(tn + stride) * TILE_F, 0, 3);
      asm volatile("s_wait_asynccnt 10" ::: "memory");  // current tile resident
    } else {
      asm volatile("s_wait_asynccnt 0" ::: "memory");
    }
    const float* buf = &s_tile[wid][cur][0];
    v8f acc;
#pragma unroll
    for (int r = 0; r < 8; r++) acc[r] = 0.f;
#pragma unroll
    for (int chunk = 0; chunk < 3; chunk++) {
      int cb = chunk * 32;
      v16h av;
#pragma unroll
      for (int e = 0; e < 16; e++) {
        // ISA 16-bit A 16x32 layout: lanes0-15 K=e(+8 for e>=8); lanes16-31 +8
        int K = e + 8*half + ((e >= 8) ? 8 : 0);
        int c = cb + K;
        float val = 0.f;
        if (c < C_) {
          float p = fminf(fmaxf(buf[M*C_ + c], 1e-7f), 1.f - 1e-7f);
          val = -logf(1.f - p);
        }
        av[e] = (_Float16)val;
      }
      acc = __builtin_amdgcn_wmma_f32_16x16x32_f16(false, av, false, ones,
                                                   (short)0, acc, false, false);
    }
    if (M == 0) {
#pragma unroll
      for (int r = 0; r < 8; r++) S[t*16 + r + 8*half] = acc[r];  // D: lane0 rows0-7, lane16 rows8-15
    }
    cur ^= 1;
  }
}

// ---------- K4: fg finalize (cls adjust, CIoU, DFL) + OHEM array ----------
__global__ void k_finalize(const float* __restrict__ cls,
                           const float* __restrict__ reg,
                           const float4* __restrict__ boxes,
                           const float* __restrict__ gtb,
                           const int* __restrict__ gtl,
                           const float* __restrict__ ap,
                           const float* __restrict__ st,
                           const unsigned long long* __restrict__ keys,
                           const unsigned* __restrict__ fg,
                           float* __restrict__ S,
                           int* __restrict__ npos, float* __restrict__ pos,
                           float* __restrict__ boxs, float* __restrict__ dfls) {
  int idx = blockIdx.x * blockDim.x + threadIdx.x;
  if (idx >= B_ * A_) return;
  int b = idx / A_, a = idx % A_;
  float Sv = S[idx];
  if (fg[idx] == 0u) return;  // OHEM value stays = Sv
  unsigned long long key = keys[idx];
  int g = 31 - (int)(key & 0xffffffffULL);
  if (g < 0) g = 0; if (g > 31) g = 31;
  int t = gtl[b*G_ + g];
  float pt = fminf(fmaxf(cls[(size_t)idx*C_ + t], 1e-7f), 1.f - 1e-7f);
  float per = Sv + logf(1.f - pt) - logf(pt);  // swap BCE term of target class
  const float* gb = gtb + ((size_t)b*G_ + g)*4;
  float tx1 = gb[0], ty1 = gb[1], tx2 = gb[2], ty2 = gb[3];
  float4 p = boxes[idx];
  float pw = fmaxf(p.z-p.x, 0.f), ph = fmaxf(p.w-p.y, 0.f);
  float gw = fmaxf(tx2-tx1, 0.f), gh = fmaxf(ty2-ty1, 0.f);
  float pcx = (p.x+p.z)*0.5f, pcy = (p.y+p.w)*0.5f;
  float gcx = (tx1+tx2)*0.5f, gcy = (ty1+ty2)*0.5f;
  float iw = fmaxf(fminf(p.z, tx2) - fmaxf(p.x, tx1), 0.f);
  float ih = fmaxf(fminf(p.w, ty2) - fmaxf(p.y, ty1), 0.f);
  float inter = iw * ih;
  float un = pw*ph + gw*gh - inter + 1e-9f;
  float iou = inter / un;
  float cw = fmaxf(fmaxf(p.z, tx2) - fminf(p.x, tx1), 0.f);
  float ch = fmaxf(fmaxf(p.w, ty2) - fminf(p.y, ty1), 0.f);
  float c2 = cw*cw + ch*ch + 1e-9f;
  float rho2 = (pcx-gcx)*(pcx-gcx) + (pcy-gcy)*(pcy-gcy);
  float dv = atanf(gw/(gh + 1e-9f)) - atanf(pw/(ph + 1e-9f));
  float v = (float)(4.0/(M_PI*M_PI)) * dv * dv;
  float alpha_v = v / (1.f - iou + v + 1e-9f);
  float ciou = 1.f - iou + rho2/c2 + alpha_v*v;
  // DFL
  float axp = ap[2*a], ayp = ap[2*a + 1], sv = st[a];
  float dist[4];
  dist[0] = (axp - tx1)/sv; dist[1] = (ayp - ty1)/sv;
  dist[2] = (tx2 - axp)/sv; dist[3] = (ty2 - ayp)/sv;
  const float* r = reg + (size_t)idx * (4*RB);
  float dfl = 0.f;
#pragma unroll
  for (int i = 0; i < 4; i++) {
    float d = fminf(fmaxf(dist[i], 0.f), (float)RMAX - 0.01f);
    int tl = (int)d;
    float wl = (float)(tl + 1) - d;
    float m = -3.4e38f;
#pragma unroll
    for (int j = 0; j < RB; j++) m = fmaxf(m, r[i*RB + j]);
    float se = 0.f;
#pragma unroll
    for (int j = 0; j < RB; j++) se += expf(r[i*RB + j] - m);
    float lse = logf(se) + m;
    float lpl = r[i*RB + tl] - lse;
    float lpr = r[i*RB + tl + 1] - lse;
    dfl += (-lpl)*wl + (-lpr)*(1.f - wl);
  }
  dfl *= 0.25f;
  atomicAdd(&npos[b], 1);
  atomicAdd(&pos[b], per);
  atomicAdd(&boxs[b], ciou);
  atomicAdd(&dfls[b], dfl);
  S[idx] = -1e30f;  // exclude fg from OHEM
}

// ---------- K5: OHEM sum of top num_hard via 3-level histogram select ----------
__global__ void k_ohem(const float* __restrict__ S, const int* __restrict__ npos,
                       float* __restrict__ negsum) {
  __shared__ int hist[1024];
  __shared__ float sh_lo, sh_hi, sh_sum, sh_newhi, sh_psum;
  __shared__ int sh_cnt, sh_cumabove;
  int b = blockIdx.x, tid = threadIdx.x;
  const float* nv = S + (size_t)b * A_;
  int np = npos[b];
  int mh = (np > 1 ? np : 1);
  long long Kl = (long long)(A_ - np), Kh = 3LL * (long long)mh;
  int K = (int)(Kl < Kh ? Kl : Kh);
  if (K <= 0) { if (tid == 0) negsum[b] = 0.f; return; }
  if (tid == 0) { sh_lo = 0.f; sh_hi = 1312.f; sh_sum = 0.f; sh_cnt = 0; }
  __syncthreads();
  for (int level = 0; level < 3; ++level) {
    float lo = sh_lo, hi = sh_hi;
    for (int i = tid; i < 1024; i += blockDim.x) hist[i] = 0;
    __syncthreads();
    float w = (hi - lo) * (1.f/1024.f);
    for (int a = tid; a < A_; a += blockDim.x) {
      float v = nv[a];
      if (v >= lo && v < hi) {
        int bin = (int)((v - lo) / w);
        if (bin > 1023) bin = 1023; if (bin < 0) bin = 0;
        atomicAdd(&hist[bin], 1);
      }
    }
    __syncthreads();
    if (tid == 0) {
      int need = K - sh_cnt;
      int cum = 0, tb = 0;
      for (int t = 1023; t >= 0; --t) {
        cum += hist[t];
        if (cum >= need || t == 0) { tb = t; break; }
      }
      sh_cumabove = cum - hist[tb];
      sh_newhi = lo + (float)(tb + 1) * w;
      sh_lo = lo + (float)tb * w;
      sh_psum = 0.f;
    }
    __syncthreads();
    float nhi = sh_newhi;
    float local = 0.f;
    for (int a = tid; a < A_; a += blockDim.x) {
      float v = nv[a];
      if (v >= nhi && v < hi) local += v;
    }
    atomicAdd(&sh_psum, local);
    __syncthreads();
    if (tid == 0) {
      sh_sum += sh_psum;
      sh_cnt += sh_cumabove;
      sh_hi = sh_newhi;
    }
    __syncthreads();
  }
  if (tid == 0) {
    int rem = K - sh_cnt;  // remaining values lie in a ~1e-6-wide bracket
    negsum[b] = sh_sum + (float)rem * 0.5f * (sh_lo + sh_hi);
  }
}

// ---------- K6: final combine ----------
__global__ void k_combine(const int* __restrict__ npos, const float* __restrict__ pos,
                          const float* __restrict__ boxs, const float* __restrict__ dfls,
                          const float* __restrict__ negsum, float* __restrict__ out) {
  if (threadIdx.x == 0 && blockIdx.x == 0) {
    float total = 0.f;
    for (int b = 0; b < B_; b++) {
      int np = npos[b];
      float denom = (float)(np > 1 ? np : 1);
      total += (pos[b] + negsum[b] + 2.5f*boxs[b] + 0.5f*dfls[b]) / denom;
    }
    out[0] = total;
  }
}

extern "C" void kernel_launch(void* const* d_in, const int* in_sizes, int n_in,
                              void* d_out, int out_size, void* d_ws, size_t ws_size,
                              hipStream_t stream) {
  const float* cls = (const float*)d_in[0];
  const float* reg = (const float*)d_in[1];
  const float* gtb = (const float*)d_in[2];
  const int*   gtl = (const int*)d_in[3];
  const float* ap  = (const float*)d_in[4];
  const float* st  = (const float*)d_in[5];
  float* out = (float*)d_out;
  char* ws = (char*)d_ws;
  if (ws_size < WS_NEEDED) return;
  float4* boxes = (float4*)(ws + OFF_BOXES);
  float* S = (float*)(ws + OFF_S);
  unsigned long long* keys = (unsigned long long*)(ws + OFF_KEY);
  unsigned* fg = (unsigned*)(ws + OFF_FG);
  float* acc = (float*)(ws + OFF_ACC);
  int*   npos = (int*)(acc);
  float* pos  = acc + 1*B_;
  float* boxs = acc + 2*B_;
  float* dfls = acc + 3*B_;
  float* negs = acc + 4*B_;

  size_t zwords = (WS_NEEDED - OFF_KEY) / 4;
  k_zero<<<(unsigned)((zwords + 255)/256), 256, 0, stream>>>((unsigned*)(ws + OFF_KEY), zwords);
  k_decode<<<(B_*A_ + 255)/256, 256, 0, stream>>>(reg, ap, st, boxes);
  k_assign<<<B_*G_, 256, 0, stream>>>(cls, boxes, gtb, gtl, ap, keys, fg);
  k_bcesum<<<512, 256, 0, stream>>>(cls, S);
  k_finalize<<<(B_*A_ + 255)/256, 256, 0, stream>>>(cls, reg, boxes, gtb, gtl, ap, st,
                                                   keys, fg, S, npos, pos, boxs, dfls);
  k_ohem<<<B_, 256, 0, stream>>>(S, npos, negs);
  k_combine<<<1, 32, 0, stream>>>(npos, pos, boxs, dfls, negs, out);
}
